// NonlocalBlock_67731634258704
// MI455X (gfx1250) — matrix operand order repaired and checked by reading, not compiled, not run on hardware
//
#include <hip/hip_runtime.h>
#include <hip/hip_bf16.h>
#include <math.h>

typedef __attribute__((ext_vector_type(16))) __bf16 v16bf;
typedef __attribute__((ext_vector_type(8)))  __bf16 v8bf;
typedef __attribute__((ext_vector_type(4)))  __bf16 v4bf;
typedef __attribute__((ext_vector_type(8)))  float  v8f;

typedef __bf16 bf16;

#define HW 128
#define MPB 16384          // H*W per batch image
#define BATCH 4
#define CCH 64
#define CKCH 576

// ---------------- activation ----------------
template<int ACT> __device__ inline float act_f(float v) {
    if constexpr (ACT == 1) return v > 0.f ? v : 0.f;            // relu
    else if constexpr (ACT == 2) return 1.f / (1.f + expf(-v));  // sigmoid
    else if constexpr (ACT == 3) return v > 0.f ? v : 0.1f * v;  // leaky 0.1
    else if constexpr (ACT == 4) return 8.f * tanhf(v);          // MAX_DIST*tanh
    else return v;
}

__device__ inline v8bf zero8() {
    v8bf z;
    #pragma unroll
    for (int i = 0; i < 8; i++) z[i] = (bf16)0.f;
    return z;
}

// ---------------- f32 -> bf16 convert ----------------
__global__ void cvt_f32_bf16(const float* __restrict__ a, bf16* __restrict__ o, int n) {
    int i = blockIdx.x * 256 + threadIdx.x;
    if (i < n) o[i] = (bf16)a[i];
}

// ---------------- 3x3 weight transpose: (co,ci,t) f32 -> (t,co,ci) bf16 -------
__global__ void transpose_w3(const float* __restrict__ w, bf16* __restrict__ o) {
    int i = blockIdx.x * 256 + threadIdx.x;       // 9*64*64 = 36864
    if (i < 36864) {
        int t = i >> 12;                          // / 4096
        int r = i & 4095;                         // co*64+ci
        o[i] = (bf16)w[(size_t)r * 9 + t];
    }
}

// =============================================================================
// WMMA GEMM v3: block tile 64 Cout x 128 m, 8 waves (2 co x 4 m), 32x32 per
// wave; two 32-wide k-slices staged per barrier pair -> 8 v_wmma per wave/sync.
// W: Cout x Cin bf16 row-major (Cin % 64 == 0).  X: B x Cin x M bf16.
// grid:(M/128, ceil(Cout/64), B)
// =============================================================================
#define LDA 40
#define LDB 48
template<int ACT>
__global__ void gemm_wmma(const bf16* __restrict__ Wm, const bf16* __restrict__ X,
                          const float* __restrict__ bias, const float* __restrict__ resid,
                          float* __restrict__ Yf, bf16* __restrict__ Yb,
                          int Cout, int Cin, int M) {
    __shared__ alignas(32) bf16 As[2][64 * LDA];
    __shared__ alignas(32) bf16 Bs[2][128 * LDB];
    const int tid = threadIdx.x;
    const int coBase = blockIdx.y * 64;
    const int mBase = blockIdx.x * 128;
    const bf16* Xb = X + (size_t)blockIdx.z * (size_t)Cin * (size_t)M;
    const int wave = tid >> 5, lane = tid & 31;
    const int wr = wave & 1, wc = wave >> 1;
    const int half = lane >> 4, ln = lane & 15;

    v8f acc[2][2];
    #pragma unroll
    for (int i = 0; i < 2; i++)
        #pragma unroll
        for (int j = 0; j < 2; j++)
            #pragma unroll
            for (int v = 0; v < 8; v++) acc[i][j][v] = 0.f;

    const int aRow = tid >> 2, aK8 = (tid & 3) * 8;      // A: 64 rows x 32 k
    const int bKk = tid >> 3, bM16 = (tid & 7) * 16;     // B: 32 k rows x 128 m

    for (int k0 = 0; k0 < Cin; k0 += 64) {
        #pragma unroll
        for (int s = 0; s < 2; s++) {
            const int kk0 = k0 + s * 32;
            {   // A tile
                v8bf av = zero8();
                int co = coBase + aRow;
                if (co < Cout)
                    av = *(const v8bf*)(Wm + (size_t)co * Cin + kk0 + aK8);
                *(v8bf*)&As[s][aRow * LDA + aK8] = av;
            }
            {   // B tile transposed [m][k]
                const bf16* p = Xb + (size_t)(kk0 + bKk) * M + mBase + bM16;
                v8bf b0 = *(const v8bf*)(p);
                v8bf b1 = *(const v8bf*)(p + 8);
                #pragma unroll
                for (int u = 0; u < 8; u++) {
                    Bs[s][(bM16 + u    ) * LDB + bKk] = b0[u];
                    Bs[s][(bM16 + u + 8) * LDB + bKk] = b1[u];
                }
            }
        }
        if (k0 + 64 < Cin)
            __builtin_prefetch(Xb + (size_t)(k0 + 64 + bKk) * M + mBase + bM16, 0, 0);
        __syncthreads();

        #pragma unroll
        for (int s = 0; s < 2; s++) {
            v16bf a[2], b[2];
            #pragma unroll
            for (int i = 0; i < 2; i++) {
                int r = wr * 32 + i * 16 + ln;
                const v8bf a0 = *(const v8bf*)&As[s][r * LDA + half * 16];
                const v8bf a1 = *(const v8bf*)&As[s][r * LDA + half * 16 + 8];
                #pragma unroll
                for (int e = 0; e < 8; e++) { a[i][e] = a0[e]; a[i][8 + e] = a1[e]; }
            }
            #pragma unroll
            for (int j = 0; j < 2; j++) {
                int rm = wc * 32 + j * 16 + ln;
                b[j] = *(const v16bf*)&Bs[s][rm * LDB + half * 16];
            }
            #pragma unroll
            for (int i = 0; i < 2; i++)
                #pragma unroll
                for (int j = 0; j < 2; j++)
                    acc[i][j] = __builtin_amdgcn_wmma_f32_16x16x32_bf16(
                        false, a[i], false, b[j], (short)0, acc[i][j], false, false);
        }
        __syncthreads();
    }
    // ---- epilogue ----
    #pragma unroll
    for (int i = 0; i < 2; i++) {
        #pragma unroll
        for (int j = 0; j < 2; j++) {
            const int m = mBase + wc * 32 + j * 16 + ln;
            #pragma unroll
            for (int v = 0; v < 8; v++) {
                int co = coBase + wr * 32 + i * 16 + v + 8 * half;
                if (co < Cout) {
                    float val = acc[i][j][v];
                    if (bias) val += bias[co];
                    val = act_f<ACT>(val);
                    size_t idx = ((size_t)blockIdx.z * Cout + co) * (size_t)M + m;
                    if (resid) val += resid[idx];
                    if (Yf) Yf[idx] = val;
                    if (Yb) Yb[idx] = (bf16)val;
                }
            }
        }
    }
}

// =============================================================================
// WMMA 3x3 conv, Cin=Cout=64, pad=1, fused bias + leaky(0.1).
// W9t: transposed weights (t,co,ci) bf16.  grid:(MPB/64, 2, B).
// Both k-slices staged per tap -> 2 v_wmma per barrier pair; branchless edges.
// =============================================================================
__global__ void conv3x3_wmma(const bf16* __restrict__ W9t, const bf16* __restrict__ X,
                             const float* __restrict__ bias, bf16* __restrict__ Y) {
    __shared__ alignas(32) bf16 As[2][32 * LDA];
    __shared__ alignas(32) bf16 Bs[2][64 * LDB];
    const int tid = threadIdx.x;
    const int coBase = blockIdx.y * 32;
    const int mBase = blockIdx.x * 64;
    const int h = mBase >> 7;            // W=128, tile lies within one image row
    const int w0 = mBase & 127;
    const bf16* Xb = X + (size_t)blockIdx.z * CCH * MPB;
    const int wave = tid >> 5, lane = tid & 31;
    const int wr = wave & 1, wc = wave >> 1;
    const int half = lane >> 4, ln = lane & 15;

    const int aRow = tid >> 3, aK4 = (tid & 7) * 4;   // A: 32 rows x 32 k
    const int bKk = tid >> 3, bM8 = (tid & 7) * 8;    // B: 32 k rows x 64 m

    v8f acc = {};
    for (int t = 0; t < 9; t++) {
        const int dy = t / 3 - 1, dx = t % 3 - 1;
        const int sh = h + dy;
        const bool rowOK = (sh >= 0) && (sh < HW);
        const bf16* Wt = W9t + (size_t)t * (CCH * CCH);
        const int c0 = w0 + bM8 + dx;

        #pragma unroll
        for (int s = 0; s < 2; s++) {
            const int kk0 = s * 32;
            // A tile: contiguous bf16 from transposed weights
            *(v4bf*)&As[s][aRow * LDA + aK4] =
                *(const v4bf*)(Wt + (size_t)(coBase + aRow) * CCH + kk0 + aK4);
            // B tile: vector fast path; branchless clamped edge path
            v8bf bv;
            if (!rowOK) {
                bv = zero8();
            } else {
                const bf16* p = Xb + (size_t)(kk0 + bKk) * MPB + sh * HW;
                if (c0 >= 0 && c0 + 7 < HW) {
                    bv = *(const v8bf*)(p + c0);
                } else {
                    #pragma unroll
                    for (int u = 0; u < 8; u++) {
                        int c = c0 + u;
                        int cc = min(max(c, 0), HW - 1);   // always in-bounds load
                        bf16 tv = p[cc];
                        bv[u] = (c >= 0 && c < HW) ? tv : (bf16)0.f;
                    }
                }
            }
            #pragma unroll
            for (int u = 0; u < 8; u++)
                Bs[s][(bM8 + u) * LDB + bKk] = bv[u];
        }
        __syncthreads();
        #pragma unroll
        for (int s = 0; s < 2; s++) {
            v16bf a;
            {
                const v8bf a0 = *(const v8bf*)&As[s][(wr * 16 + ln) * LDA + half * 16];
                const v8bf a1 = *(const v8bf*)&As[s][(wr * 16 + ln) * LDA + half * 16 + 8];
                #pragma unroll
                for (int e = 0; e < 8; e++) { a[e] = a0[e]; a[8 + e] = a1[e]; }
            }
            v16bf bfr = *(const v16bf*)&Bs[s][(wc * 16 + ln) * LDB + half * 16];
            acc = __builtin_amdgcn_wmma_f32_16x16x32_bf16(false, a, false, bfr,
                                                          (short)0, acc, false, false);
        }
        __syncthreads();
    }
    const int m = mBase + wc * 16 + ln;
    #pragma unroll
    for (int v = 0; v < 8; v++) {
        int co = coBase + wr * 16 + v + 8 * half;
        float val = acc[v] + bias[co];
        val = val > 0.f ? val : 0.1f * val;
        Y[((size_t)blockIdx.z * CCH + co) * MPB + m] = (bf16)val;
    }
}

// ---------------- depthwise 7x7, pad 3, f32 in -> bf16 out (branchless) -------
__global__ void dw7_conv(const float* __restrict__ X, const float* __restrict__ Wd,
                         bf16* __restrict__ Y) {
    int m = blockIdx.x * 256 + threadIdx.x;
    int c = blockIdx.y, b = blockIdx.z;
    int hh = m >> 7, ww = m & 127;
    const float* src = X + ((size_t)b * CCH + c) * MPB;
    const float* wt = Wd + c * 49;
    float s = 0.f;
    for (int i = 0; i < 7; i++) {
        int y = hh + i - 3;
        int yc = min(max(y, 0), HW - 1);
        bool yok = (y >= 0) && (y < HW);
        #pragma unroll
        for (int j = 0; j < 7; j++) {
            int x = ww + j - 3;
            int xc = min(max(x, 0), HW - 1);
            float wv = (yok && x >= 0 && x < HW) ? wt[i * 7 + j] : 0.f;
            s += src[yc * HW + xc] * wv;
        }
    }
    Y[((size_t)b * CCH + c) * MPB + m] = (bf16)s;
}

// ---------------- depthwise 3x3 on CK channels, pad 1, act (branchless) -------
template<int ACT>
__global__ void dw3_conv(const bf16* __restrict__ X, const float* __restrict__ Wd,
                         bf16* __restrict__ Y) {
    int m = blockIdx.x * 256 + threadIdx.x;
    int c = blockIdx.y, b = blockIdx.z;
    int hh = m >> 7, ww = m & 127;
    const bf16* src = X + ((size_t)b * CKCH + c) * MPB;
    const float* wt = Wd + c * 9;
    float s = 0.f;
    #pragma unroll
    for (int i = 0; i < 3; i++) {
        int y = hh + i - 1;
        int yc = min(max(y, 0), HW - 1);
        bool yok = (y >= 0) && (y < HW);
        #pragma unroll
        for (int j = 0; j < 3; j++) {
            int x = ww + j - 1;
            int xc = min(max(x, 0), HW - 1);
            float wv = (yok && x >= 0 && x < HW) ? wt[i * 3 + j] : 0.f;
            s += (float)src[yc * HW + xc] * wv;
        }
    }
    Y[((size_t)b * CKCH + c) * MPB + m] = (bf16)act_f<ACT>(s);
}

// ---------------- deformable bilinear gather ----------------------------------
__global__ void deform_gather(const float* __restrict__ X, const float* __restrict__ Off,
                              bf16* __restrict__ Xb) {
    int m = blockIdx.x * 256 + threadIdx.x;
    int k = blockIdx.y, b = blockIdx.z;
    int hh = m >> 7, ww = m & 127;
    float dy = Off[((size_t)b * 18 + 2 * k) * MPB + m];
    float dx = Off[((size_t)b * 18 + 2 * k + 1) * MPB + m];
    float py = (float)hh + (float)(k / 3 - 1) + dy;
    float px = (float)ww + (float)(k % 3 - 1) + dx;
    float y0f = floorf(py), x0f = floorf(px);
    float wy1 = py - y0f, wy0 = 1.f - wy1;
    float wx1 = px - x0f, wx0 = 1.f - wx1;
    int y0 = (int)y0f, x0 = (int)x0f;

    int iy[2] = { y0, y0 + 1 }, ix[2] = { x0, x0 + 1 };
    float wy[2] = { wy0, wy1 }, wx[2] = { wx0, wx1 };
    int idx4[4]; float wgt4[4];
    #pragma unroll
    for (int a = 0; a < 4; a++) {
        int yy = iy[a >> 1], xx = ix[a & 1];
        bool valid = (yy >= 0) && (yy < HW) && (xx >= 0) && (xx < HW);
        int yc = min(max(yy, 0), HW - 1), xc = min(max(xx, 0), HW - 1);
        idx4[a] = yc * HW + xc;
        wgt4[a] = valid ? wy[a >> 1] * wx[a & 1] : 0.f;
    }
    for (int c = 0; c < CCH; c++) {
        const float* src = X + ((size_t)b * CCH + c) * MPB;
        float v = wgt4[0] * src[idx4[0]] + wgt4[1] * src[idx4[1]]
                + wgt4[2] * src[idx4[2]] + wgt4[3] * src[idx4[3]];
        Xb[((size_t)b * CKCH + c * 9 + k) * MPB + m] = (bf16)v;
    }
}

// ---------------- grouped 9-in/9-out 1x1 conv (dct / inv_w1) ------------------
template<bool PREMUL>
__global__ void grouped9_conv(const bf16* __restrict__ In, const bf16* __restrict__ In2,
                              const float* __restrict__ Wg, bf16* __restrict__ Out) {
    int m = blockIdx.x * 256 + threadIdx.x;
    int g = blockIdx.y, b = blockIdx.z;
    float in[9];
    #pragma unroll
    for (int i = 0; i < 9; i++) {
        size_t idx = ((size_t)b * CKCH + g * 9 + i) * MPB + m;
        float v = (float)In[idx];
        if (PREMUL) v *= (float)In2[idx];
        in[i] = v;
    }
    #pragma unroll
    for (int j = 0; j < 9; j++) {
        const float* w = Wg + (size_t)(g * 9 + j) * 9;
        float s = 0.f;
        #pragma unroll
        for (int i = 0; i < 9; i++) s += w[i] * in[i];
        Out[((size_t)b * CKCH + g * 9 + j) * MPB + m] = (bf16)s;
    }
}

// =============================================================================
extern "C" void kernel_launch(void* const* d_in, const int* in_sizes, int n_in,
                              void* d_out, int out_size, void* d_ws, size_t ws_size,
                              hipStream_t stream) {
    const float* x       = (const float*)d_in[0];
    const float* enc_dw  = (const float*)d_in[1];
    const float* enc_pw1 = (const float*)d_in[2];
    const float* enc_pw2 = (const float*)d_in[3];
    const float* dec_dw  = (const float*)d_in[4];
    const float* dec_pw1 = (const float*)d_in[5];
    const float* dec_pw2 = (const float*)d_in[6];
    const float* off_w1 = (const float*)d_in[7];  const float* off_b1 = (const float*)d_in[8];
    const float* off_w2 = (const float*)d_in[9];  const float* off_b2 = (const float*)d_in[10];
    const float* off_w3 = (const float*)d_in[11]; const float* off_b3 = (const float*)d_in[12];
    const float* off_w4 = (const float*)d_in[13]; const float* off_b4 = (const float*)d_in[14];
    const float* off_w5 = (const float*)d_in[15]; const float* off_b5 = (const float*)d_in[16];
    const float* off_w6 = (const float*)d_in[17]; const float* off_b6 = (const float*)d_in[18];
    const float* dct_w  = (const float*)d_in[19];
    const float* wie_w1 = (const float*)d_in[20]; const float* wie_w2 = (const float*)d_in[21];
    const float* wie_w3 = (const float*)d_in[22]; const float* wie_w4 = (const float*)d_in[23];
    const float* wie_w5 = (const float*)d_in[24]; const float* wie_w6 = (const float*)d_in[25];
    const float* inv_w1 = (const float*)d_in[26];
    const float* inv_w2 = (const float*)d_in[27];

    const int Mv = MPB;
    char* base = (char*)d_ws;
    size_t cur = 0;
    auto alloc = [&](size_t bytes) -> void* {
        void* p = base + cur;
        cur = (cur + bytes + 255) & ~(size_t)255;
        return p;
    };
    const size_t BM = (size_t)BATCH * Mv;

    // bf16 weight copies
    bf16* wb_epw1 = (bf16*)alloc(8192u * 2);
    bf16* wb_epw2 = (bf16*)alloc(8192u * 2);
    bf16* wb_dpw1 = (bf16*)alloc(8192u * 2);
    bf16* wb_dpw2 = (bf16*)alloc(8192u * 2);
    bf16* wb_o1   = (bf16*)alloc(4096u * 2);
    bf16* wb_o2   = (bf16*)alloc(36864u * 2);   // transposed (t,co,ci)
    bf16* wb_o3   = (bf16*)alloc(36864u * 2);
    bf16* wb_o4   = (bf16*)alloc(36864u * 2);
    bf16* wb_o5   = (bf16*)alloc(36864u * 2);
    bf16* wb_o6   = (bf16*)alloc(1152u * 2);
    bf16* wb_w1   = (bf16*)alloc(331776u * 2);
    bf16* wb_w3   = (bf16*)alloc(331776u * 2);
    bf16* wb_w5   = (bf16*)alloc(331776u * 2);
    bf16* wb_i2   = (bf16*)alloc(36864u * 2);
    // activations
    float* x1f  = (float*)alloc(64 * BM * 4);
    bf16*  x1b  = (bf16*) alloc(64 * BM * 2);
    bf16*  t128 = (bf16*) alloc(128 * BM * 2);
    bf16*  offA = (bf16*) alloc(64 * BM * 2);
    bf16*  offB = (bf16*) alloc(64 * BM * 2);
    float* offs = (float*)alloc(18 * BM * 4);
    bf16*  xb   = (bf16*) alloc((size_t)CKCH * BM * 2);
    bf16*  dctb = (bf16*) alloc((size_t)CKCH * BM * 2);
    bf16*  yA   = (bf16*) alloc((size_t)CKCH * BM * 2);
    bf16*  yB   = (bf16*) alloc((size_t)CKCH * BM * 2);
    float* out0 = (float*)alloc(64 * BM * 4);
    bf16*  dwt  = (bf16*) alloc(64 * BM * 2);

    auto cvt = [&](const float* s, bf16* d, int n) {
        cvt_f32_bf16<<<dim3((n + 255) / 256), dim3(256), 0, stream>>>(s, d, n);
    };
    cvt(enc_pw1, wb_epw1, 8192);  cvt(enc_pw2, wb_epw2, 8192);
    cvt(dec_pw1, wb_dpw1, 8192);  cvt(dec_pw2, wb_dpw2, 8192);
    cvt(off_w1, wb_o1, 4096);     cvt(off_w6, wb_o6, 1152);
    cvt(wie_w1, wb_w1, 331776);   cvt(wie_w3, wb_w3, 331776);
    cvt(wie_w5, wb_w5, 331776);   cvt(inv_w2, wb_i2, 36864);
    dim3 gT((36864 + 255) / 256), blk(256);
    transpose_w3<<<gT, blk, 0, stream>>>(off_w2, wb_o2);
    transpose_w3<<<gT, blk, 0, stream>>>(off_w3, wb_o3);
    transpose_w3<<<gT, blk, 0, stream>>>(off_w4, wb_o4);
    transpose_w3<<<gT, blk, 0, stream>>>(off_w5, wb_o5);

    auto gg = [&](int Cout) { return dim3(Mv / 128, (Cout + 63) / 64, BATCH); };
    dim3 gPix(Mv / 256, CCH, BATCH);
    dim3 gC3(Mv / 64, 2, BATCH);

    // ---- encoder ConvNeXt ----
    dw7_conv<<<gPix, blk, 0, stream>>>(x, enc_dw, dwt);
    gemm_wmma<1><<<gg(128), blk, 0, stream>>>(wb_epw1, dwt, nullptr, nullptr, nullptr, t128, 128, 64, Mv);
    gemm_wmma<0><<<gg(64), blk, 0, stream>>>(wb_epw2, t128, nullptr, x, x1f, x1b, 64, 128, Mv);

    // ---- offset CNN ----
    gemm_wmma<3><<<gg(64), blk, 0, stream>>>(wb_o1, x1b, off_b1, nullptr, nullptr, offA, 64, 64, Mv);
    conv3x3_wmma<<<gC3, blk, 0, stream>>>(wb_o2, offA, off_b2, offB);
    conv3x3_wmma<<<gC3, blk, 0, stream>>>(wb_o3, offB, off_b3, offA);
    conv3x3_wmma<<<gC3, blk, 0, stream>>>(wb_o4, offA, off_b4, offB);
    conv3x3_wmma<<<gC3, blk, 0, stream>>>(wb_o5, offB, off_b5, offA);
    gemm_wmma<4><<<gg(18), blk, 0, stream>>>(wb_o6, offA, off_b6, nullptr, offs, nullptr, 18, 64, Mv);

    // ---- deformable gather + dct ----
    deform_gather<<<dim3(Mv / 256, 9, BATCH), blk, 0, stream>>>(x1f, offs, xb);
    grouped9_conv<false><<<dim3(Mv / 256, 64, BATCH), blk, 0, stream>>>(xb, nullptr, dct_w, dctb);

    // ---- Wiener chain: 3 big 576x576 WMMA GEMMs + depthwise 3x3 ----
    gemm_wmma<0><<<gg(CKCH), blk, 0, stream>>>(wb_w1, xb, nullptr, nullptr, nullptr, yA, CKCH, CKCH, Mv);
    dw3_conv<1><<<dim3(Mv / 256, CKCH, BATCH), blk, 0, stream>>>(yA, wie_w2, yB);
    gemm_wmma<0><<<gg(CKCH), blk, 0, stream>>>(wb_w3, yB, nullptr, nullptr, nullptr, yA, CKCH, CKCH, Mv);
    dw3_conv<1><<<dim3(Mv / 256, CKCH, BATCH), blk, 0, stream>>>(yA, wie_w4, yB);
    gemm_wmma<0><<<gg(CKCH), blk, 0, stream>>>(wb_w5, yB, nullptr, nullptr, nullptr, yA, CKCH, CKCH, Mv);
    dw3_conv<2><<<dim3(Mv / 256, CKCH, BATCH), blk, 0, stream>>>(yA, wie_w6, yB);  // sigmoid

    // ---- inverse: (wiener*dct) grouped conv, then 64<-576 GEMM (xb reused) ----
    grouped9_conv<true><<<dim3(Mv / 256, 64, BATCH), blk, 0, stream>>>(yB, dctb, inv_w1, xb);
    gemm_wmma<0><<<gg(64), blk, 0, stream>>>(wb_i2, xb, nullptr, nullptr, out0, nullptr, 64, CKCH, Mv);

    // ---- decoder ConvNeXt ----
    dw7_conv<<<gPix, blk, 0, stream>>>(out0, dec_dw, dwt);
    gemm_wmma<1><<<gg(128), blk, 0, stream>>>(wb_dpw1, dwt, nullptr, nullptr, nullptr, t128, 128, 64, Mv);
    gemm_wmma<0><<<gg(64), blk, 0, stream>>>(wb_dpw2, t128, nullptr, out0, (float*)d_out, nullptr, 64, 128, Mv);

    (void)in_sizes; (void)n_in; (void)out_size; (void)ws_size;
}
